// MultiHeadAttention_16028817949051
// MI455X (gfx1250) — matrix operand order repaired and checked by reading, not compiled, not run on hardware
//
#include <hip/hip_runtime.h>

#define CC 512
#define TT 1024
#define BB 16
#define NH 8
#define DH 64

typedef __attribute__((ext_vector_type(16))) __bf16 v16bf;
typedef __attribute__((ext_vector_type(8)))  __bf16 v8bf;
typedef __attribute__((ext_vector_type(4)))  __bf16 v4bf;
typedef __attribute__((ext_vector_type(8)))  float  v8f;
typedef int v4i_gcc __attribute__((vector_size(16)));

#if defined(__has_builtin)
#if __has_builtin(__builtin_amdgcn_global_load_async_to_lds_b128) && \
    __has_builtin(__builtin_amdgcn_s_wait_asynccnt)
#define HAVE_ASYNC_LDS 1
#endif
#endif

static __device__ __forceinline__ v8f wmma_bf16(v16bf a, v16bf b, v8f c) {
  // D = A(16x32 bf16) * B(32x16 bf16) + C(16x16 f32)
  return __builtin_amdgcn_wmma_f32_16x16x32_bf16(false, a, false, b, (short)0, c,
                                                 false, false);
}

static __device__ __forceinline__ v16bf packA(v8bf lo, v8bf hi) {
  v16bf r;
#pragma unroll
  for (int i = 0; i < 8; ++i) { r[i] = lo[i]; r[i + 8] = hi[i]; }
  return r;
}

// 16-byte global->LDS copy, async if the toolchain exposes the gfx1250 builtin.
static __device__ __forceinline__ void cp_g2l_16B(const __bf16* g, __bf16* l) {
#ifdef HAVE_ASYNC_LDS
  __builtin_amdgcn_global_load_async_to_lds_b128(
      (__attribute__((address_space(1))) v4i_gcc*)(uintptr_t)g,
      (__attribute__((address_space(3))) v4i_gcc*)(uint32_t)(uintptr_t)l, 0, 0);
#else
  *(v8bf*)l = *(const v8bf*)g;
#endif
}

static __device__ __forceinline__ void wait_g2l() {
#ifdef HAVE_ASYNC_LDS
  __builtin_amdgcn_s_wait_asynccnt(0);
#endif
}

// ---------------- W (C,C) f32 -> bf16 ----------------
__global__ __launch_bounds__(256) void w2bf_kernel(const float* __restrict__ W,
                                                   __bf16* __restrict__ Wb) {
  const int i = blockIdx.x * 256 + threadIdx.x;  // CC*CC/4 threads
  const float4 f = ((const float4*)W)[i];
  v4bf r;
  r[0] = (__bf16)f.x; r[1] = (__bf16)f.y; r[2] = (__bf16)f.z; r[3] = (__bf16)f.w;
  ((v4bf*)Wb)[i] = r;
}

// ---------------- x (B,C,T) f32 -> xT (B,T,C) bf16 ----------------
__global__ __launch_bounds__(256) void transpose_kernel(const float* __restrict__ x,
                                                        __bf16* __restrict__ xT) {
  __shared__ float tile[32][33];
  const int b = blockIdx.z;
  const int t0 = blockIdx.x * 32, c0 = blockIdx.y * 32;
  const int tx = threadIdx.x, ty = threadIdx.y;
  const float* xp = x + ((size_t)b * CC + c0) * TT + t0;
#pragma unroll
  for (int j = 0; j < 4; ++j)
    tile[ty + 8 * j][tx] = xp[(size_t)(ty + 8 * j) * TT + tx];
  __syncthreads();
  __bf16* op = xT + ((size_t)b * TT + t0) * CC + c0;
#pragma unroll
  for (int j = 0; j < 4; ++j)
    op[(size_t)(ty + 8 * j) * CC + tx] = (__bf16)tile[tx][ty + 8 * j];
}

// ---------------- fused QKV projection ----------------
__global__ __launch_bounds__(128) void qkv_proj_kernel(
    const __bf16* __restrict__ xT,
    const __bf16* __restrict__ Wq, const float* __restrict__ bq,
    const __bf16* __restrict__ Wk, const float* __restrict__ bk,
    const __bf16* __restrict__ Wv, const float* __restrict__ bv,
    __bf16* __restrict__ qo, __bf16* __restrict__ ko, __bf16* __restrict__ vo) {
  const int lane = threadIdx.x & 31;
  const int w = threadIdx.x >> 5;
  const int ln = lane & 15, lh = lane >> 4;
  const int b = blockIdx.z;
  const int m0 = blockIdx.y * 64 + w * 16;
  const int n0 = blockIdx.x * 64;
  const int rowA = m0 + ln;

  v8f accQ[4] = {}, accK[4] = {}, accV[4] = {};

  for (int k0 = 0; k0 < CC; k0 += 32) {
    v16bf bx[4];
#pragma unroll
    for (int j = 0; j < 4; ++j) {
      const int t = n0 + 16 * j + ln;
      bx[j] = packA(*(const v8bf*)(xT + ((size_t)(b * TT + t)) * CC + k0 + lh * 8), *(const v8bf*)(xT + ((size_t)(b * TT + t)) * CC + k0 + lh * 8 + 16));
    }
    const size_t aoff = (size_t)rowA * CC + k0 + lh * 8;
    v16bf aQ = packA(*(const v8bf*)(Wq + aoff), *(const v8bf*)(Wq + aoff + 16));
    v16bf aK = packA(*(const v8bf*)(Wk + aoff), *(const v8bf*)(Wk + aoff + 16));
    v16bf aV = packA(*(const v8bf*)(Wv + aoff), *(const v8bf*)(Wv + aoff + 16));
#pragma unroll
    for (int j = 0; j < 4; ++j) {
      accQ[j] = wmma_bf16(aQ, bx[j], accQ[j]);
      accK[j] = wmma_bf16(aK, bx[j], accK[j]);
      accV[j] = wmma_bf16(aV, bx[j], accV[j]);
    }
  }

#pragma unroll
  for (int r = 0; r < 8; ++r) {
    const int o = m0 + r + 8 * lh;
    const int hh = o >> 6, d = o & 63;
    const float bQ = bq[o], bK = bk[o], bV = bv[o];
#pragma unroll
    for (int j = 0; j < 4; ++j) {
      const int t = n0 + 16 * j + ln;
      const size_t qi = ((size_t)((b * NH + hh) * TT + t)) * DH + d;
      const size_t vi = ((size_t)((b * NH + hh) * DH + d)) * TT + t;
      qo[qi] = (__bf16)(accQ[j][r] + bQ);
      ko[qi] = (__bf16)(accK[j][r] + bK);
      vo[vi] = (__bf16)(accV[j][r] + bV);
    }
  }
}

// ---------------- RoPE in place on (B,H,T,D) bf16, first 32 dims ----------------
__global__ __launch_bounds__(256) void rope_kernel(__bf16* __restrict__ buf) {
  const int idx = blockIdx.x * 256 + threadIdx.x;
  const int i = idx & 15;
  const int t = (idx >> 4) & (TT - 1);
  const int bh = idx >> 14;
  __bf16* p = buf + ((size_t)bh * TT + t) * DH;
  const float a = (float)p[i];
  const float bb = (float)p[i + 16];
  const float theta = __powf(10000.0f, -(float)i * (1.0f / 16.0f));
  const float ang = (float)t * theta;
  const float s = __sinf(ang), c = __cosf(ang);
  p[i] = (__bf16)(a * c - bb * s);
  p[i + 16] = (__bf16)(bb * c + a * s);
}

// ---------------- flash attention ----------------
__global__ __launch_bounds__(256) void attn_kernel(
    const __bf16* __restrict__ q, const __bf16* __restrict__ k,
    const __bf16* __restrict__ vT, __bf16* __restrict__ att) {
  __shared__ __align__(16) __bf16 Kbuf[2][32][72];
  __shared__ __align__(16) __bf16 Vbuf[2][64][40];
  __shared__ __align__(16) __bf16 Plds[8][16][40];
  const int tid = threadIdx.x;
  const int lane = tid & 31;
  const int w = tid >> 5;
  const int ln = lane & 15, lh = lane >> 4;
  const int b = blockIdx.z, h = blockIdx.y;
  const int tq0 = blockIdx.x * 128 + w * 16;

  const __bf16* qb = q + (size_t)((b * NH + h) * TT) * DH;
  const __bf16* kp = k + (size_t)((b * NH + h) * TT) * DH;
  const __bf16* vb = vT + (size_t)((b * NH + h) * DH) * TT;

  const int krow = tid >> 3, kcol = (tid & 7) * 8;  // 32 x 64
  const int vrow = tid >> 2, vcol = (tid & 3) * 8;  // 64 x 32

  v16bf aq[2];
#pragma unroll
  for (int s = 0; s < 2; ++s) {
    const __bf16* p = qb + (size_t)(tq0 + ln) * DH + 32 * s + lh * 8;
    aq[s] = packA(*(const v8bf*)p, *(const v8bf*)(p + 16));
  }

  v8f O0 = {}, O1 = {}, O2 = {}, O3 = {};
  float rowmax[8], rowsum[8];
#pragma unroll
  for (int r = 0; r < 8; ++r) { rowmax[r] = -1e30f; rowsum[r] = 0.0f; }

  cp_g2l_16B(kp + (size_t)krow * DH + kcol, &Kbuf[0][krow][kcol]);
  cp_g2l_16B(vb + (size_t)vrow * TT + vcol, &Vbuf[0][vrow][vcol]);

  const float sc = 0.125f;
  for (int it = 0; it < TT / 32; ++it) {
    const int cur = it & 1;
    const int kb = it * 32;
    wait_g2l();
    __syncthreads();
    if (it + 1 < TT / 32) {
      const int kn = kb + 32;
      cp_g2l_16B(kp + (size_t)(kn + krow) * DH + kcol, &Kbuf[cur ^ 1][krow][kcol]);
      cp_g2l_16B(vb + (size_t)vrow * TT + kn + vcol, &Vbuf[cur ^ 1][vrow][vcol]);
    }

    v8f S0 = {}, S1 = {};
#pragma unroll
    for (int s = 0; s < 2; ++s) {
      const __bf16* r0 = &Kbuf[cur][ln][32 * s + 8 * lh];
      const __bf16* r1 = &Kbuf[cur][16 + ln][32 * s + 8 * lh];
      v16bf b0 = packA(*(const v8bf*)r0, *(const v8bf*)(r0 + 16));
      v16bf b1 = packA(*(const v8bf*)r1, *(const v8bf*)(r1 + 16));
      S0 = wmma_bf16(aq[s], b0, S0);
      S1 = wmma_bf16(aq[s], b1, S1);
    }

    float p0a[8], p1a[8];
#pragma unroll
    for (int r = 0; r < 8; ++r) {
      const float s0 = S0[r] * sc, s1 = S1[r] * sc;
      float m = fmaxf(s0, s1);
      m = fmaxf(m, __shfl_xor(m, 1, 32));
      m = fmaxf(m, __shfl_xor(m, 2, 32));
      m = fmaxf(m, __shfl_xor(m, 4, 32));
      m = fmaxf(m, __shfl_xor(m, 8, 32));
      const float newm = fmaxf(rowmax[r], m);
      const float corr = __expf(rowmax[r] - newm);
      rowmax[r] = newm;
      const float p0 = __expf(s0 - newm), p1 = __expf(s1 - newm);
      p0a[r] = p0; p1a[r] = p1;
      float rs = p0 + p1;
      rs += __shfl_xor(rs, 1, 32);
      rs += __shfl_xor(rs, 2, 32);
      rs += __shfl_xor(rs, 4, 32);
      rs += __shfl_xor(rs, 8, 32);
      rowsum[r] = rowsum[r] * corr + rs;
      O0[r] *= corr; O1[r] *= corr; O2[r] *= corr; O3[r] *= corr;
    }

#pragma unroll
    for (int r = 0; r < 8; ++r) {
      Plds[w][r + 8 * lh][ln] = (__bf16)p0a[r];
      Plds[w][r + 8 * lh][16 + ln] = (__bf16)p1a[r];
    }
    v16bf aP = packA(*(const v8bf*)&Plds[w][ln][lh * 8],
                     *(const v8bf*)&Plds[w][ln][16 + lh * 8]);

#pragma unroll
    for (int j = 0; j < 4; ++j) {
      const __bf16* rv = &Vbuf[cur][16 * j + ln][8 * lh];
      v16bf bvf = packA(*(const v8bf*)rv, *(const v8bf*)(rv + 16));
      if (j == 0) O0 = wmma_bf16(aP, bvf, O0);
      else if (j == 1) O1 = wmma_bf16(aP, bvf, O1);
      else if (j == 2) O2 = wmma_bf16(aP, bvf, O2);
      else O3 = wmma_bf16(aP, bvf, O3);
    }
  }

#pragma unroll
  for (int r = 0; r < 8; ++r) {
    const float inv = 1.0f / rowsum[r];
    const int t = tq0 + r + 8 * lh;
    __bf16* ob = att + (size_t)(b * TT + t) * CC + h * DH + ln;
    ob[0] = (__bf16)(O0[r] * inv);
    ob[16] = (__bf16)(O1[r] * inv);
    ob[32] = (__bf16)(O2[r] * inv);
    ob[48] = (__bf16)(O3[r] * inv);
  }
}

// ---------------- output projection ----------------
__global__ __launch_bounds__(128) void out_proj_kernel(
    const __bf16* __restrict__ attT, const __bf16* __restrict__ Wo,
    const float* __restrict__ bo, float* __restrict__ y) {
  const int lane = threadIdx.x & 31;
  const int w = threadIdx.x >> 5;
  const int ln = lane & 15, lh = lane >> 4;
  const int b = blockIdx.z;
  const int m0 = blockIdx.y * 64 + w * 16;
  const int n0 = blockIdx.x * 64;
  const int rowA = m0 + ln;

  v8f acc[4] = {};
  for (int k0 = 0; k0 < CC; k0 += 32) {
    v16bf bx[4];
#pragma unroll
    for (int j = 0; j < 4; ++j) {
      const int t = n0 + 16 * j + ln;
      bx[j] = packA(*(const v8bf*)(attT + ((size_t)(b * TT + t)) * CC + k0 + lh * 8), *(const v8bf*)(attT + ((size_t)(b * TT + t)) * CC + k0 + lh * 8 + 16));
    }
    const size_t aoff = (size_t)rowA * CC + k0 + lh * 8;
    v16bf aW = packA(*(const v8bf*)(Wo + aoff), *(const v8bf*)(Wo + aoff + 16));
#pragma unroll
    for (int j = 0; j < 4; ++j) acc[j] = wmma_bf16(aW, bx[j], acc[j]);
  }

#pragma unroll
  for (int r = 0; r < 8; ++r) {
    const int o = m0 + r + 8 * lh;
    const float bO = bo[o];
#pragma unroll
    for (int j = 0; j < 4; ++j) {
      const int t = n0 + 16 * j + ln;
      y[((size_t)(b * CC + o)) * TT + t] = acc[j][r] + bO;
    }
  }
}

extern "C" void kernel_launch(void* const* d_in, const int* in_sizes, int n_in,
                              void* d_out, int out_size, void* d_ws, size_t ws_size,
                              hipStream_t stream) {
  const float* x = (const float*)d_in[0];
  const float* Wq = (const float*)d_in[1];
  const float* bq = (const float*)d_in[2];
  const float* Wk = (const float*)d_in[3];
  const float* bk = (const float*)d_in[4];
  const float* Wv = (const float*)d_in[5];
  const float* bv = (const float*)d_in[6];
  const float* Wo = (const float*)d_in[7];
  const float* bo = (const float*)d_in[8];
  float* y = (float*)d_out;

  char* ws = (char*)d_ws;
  const size_t seg = (size_t)BB * TT * CC * sizeof(__bf16);  // 16 MB
  const size_t wseg = (size_t)CC * CC * sizeof(__bf16);      // 512 KB
  __bf16* xT = (__bf16*)(ws);
  __bf16* qb = (__bf16*)(ws + seg);
  __bf16* kb = (__bf16*)(ws + 2 * seg);
  __bf16* vb = (__bf16*)(ws + 3 * seg);
  __bf16* att = (__bf16*)(ws + 4 * seg);
  __bf16* Wqb = (__bf16*)(ws + 5 * seg);
  __bf16* Wkb = (__bf16*)(ws + 5 * seg + wseg);
  __bf16* Wvb = (__bf16*)(ws + 5 * seg + 2 * wseg);
  __bf16* Wob = (__bf16*)(ws + 5 * seg + 3 * wseg);  // 82 MB total

  const int wblk = (CC * CC / 4) / 256;
  w2bf_kernel<<<wblk, 256, 0, stream>>>(Wq, Wqb);
  w2bf_kernel<<<wblk, 256, 0, stream>>>(Wk, Wkb);
  w2bf_kernel<<<wblk, 256, 0, stream>>>(Wv, Wvb);
  w2bf_kernel<<<wblk, 256, 0, stream>>>(Wo, Wob);
  transpose_kernel<<<dim3(TT / 32, CC / 32, BB), dim3(32, 8), 0, stream>>>(x, xT);
  qkv_proj_kernel<<<dim3(TT / 64, CC / 64, BB), 128, 0, stream>>>(
      xT, Wqb, bq, Wkb, bk, Wvb, bv, qb, kb, vb);
  const int nrope = BB * NH * TT * 16;
  rope_kernel<<<nrope / 256, 256, 0, stream>>>(qb);
  rope_kernel<<<nrope / 256, 256, 0, stream>>>(kb);
  attn_kernel<<<dim3(TT / 128, NH, BB), 256, 0, stream>>>(qb, kb, vb, att);
  out_proj_kernel<<<dim3(TT / 64, CC / 64, BB), 128, 0, stream>>>(att, Wob, bo, y);
}
